// Model_18150531793320
// MI455X (gfx1250) — matrix operand order repaired and checked by reading, not compiled, not run on hardware
//
#include <hip/hip_runtime.h>
#include <hip/hip_bf16.h>

typedef long long i64;
typedef __attribute__((ext_vector_type(16))) __bf16 v16bf;
typedef __attribute__((ext_vector_type(8)))  __bf16 v8bf;
typedef __attribute__((ext_vector_type(8)))  float  v8f;

#define NN   8192
#define BZ   4
#define HID  128
#define NH   4
#define HD   512   // NH * 128

// ---------------- WMMA fragment helpers (16x16x32 bf16, contiguous k order used for both A and B) ----------------
// A matrix 16x32 bf16, lane L: half=L>>4, row=L&15.
// Lane's 16 elements = A[row, half*16 .. +15]  (one 32B run)
__device__ inline v16bf load_frag_a(const __bf16* __restrict__ A, int lda) {
  int lane = threadIdx.x & 31;
  int half = lane >> 4, row = lane & 15;
  return *(const v16bf*)(A + (i64)row * lda + half * 16);
}
// B matrix 32x16 bf16, lane L: half=L>>4, col=L&15; b[i] = B[half*16+i, col].
// With B stored TRANSPOSED (BT[N,K] row-major): lane's 16 elements = BT[col, half*16 .. +15]
__device__ inline v16bf load_frag_bT(const __bf16* __restrict__ BT, int ldbt) {
  int lane = threadIdx.x & 31;
  int half = lane >> 4, col = lane & 15;
  return *(const v16bf*)(BT + (i64)col * ldbt + half * 16);
}
// C/D 16x16 f32: lane L: col=L&15; VGPR r -> row = r + 8*(L>>4)
__device__ inline void store_frag_c(float* __restrict__ C, int ldc, v8f c,
                                    const float* __restrict__ bias, int colBase) {
  int lane = threadIdx.x & 31;
  int half = lane >> 4, col = lane & 15;
  float badd = bias ? bias[colBase + col] : 0.0f;
#pragma unroll
  for (int r = 0; r < 8; ++r)
    C[(i64)(r + half * 8) * ldc + col] = c[r] + badd;
}

// ---------------- Generic batched bf16 GEMM, f32 accumulate ----------------
// C[M,N] = A[M,K] @ B[K,N] (+ bias[col]); B given TRANSPOSED as BT[N,K] row-major.
// Block = 128 threads = 4 waves; wave tile = 16 rows x (JF*16) cols.
// Software-pipelined ping-pong: fragments for step s+1 / s+2 are loaded while the
// wmma chain for step s executes, so waits are partial instead of loadcnt==0.
// Requirements (all launches obey): M%64==0, N%(JF*16)==0, K%64==0, ldbt==K.
// NOTE: the final-iteration preload overreads one 64B k-step past A/BT; the
// workspace layout appends a pad region so this always lands in mapped scratch.
template <int JF>
__global__ __launch_bounds__(128) void gemm_bf16(
    const __bf16* __restrict__ A, const __bf16* __restrict__ BT,
    float* __restrict__ C, const float* __restrict__ bias,
    int M, int N, int K, int lda, int ldbt, int ldc,
    i64 sA, i64 sB, i64 sC) {
  i64 z = blockIdx.z;
  A += z * sA; BT += z * sB; C += z * sC;
  int wave = threadIdx.x >> 5;
  int row0 = blockIdx.x * 64 + wave * 16;
  int col0 = blockIdx.y * (JF * 16);
  v8f acc[JF];
#pragma unroll
  for (int j = 0; j < JF; ++j) acc[j] = v8f{};
  const __bf16* Arow  = A + (i64)row0 * lda;
  const __bf16* Bbase = BT + (i64)col0 * ldbt;

  v16bf a0 = load_frag_a(Arow, lda);
  v16bf b0[JF];
#pragma unroll
  for (int j = 0; j < JF; ++j) b0[j] = load_frag_bT(Bbase + (i64)j * 16 * ldbt, ldbt);

  int steps = K >> 5;  // even for all launches (K=128 or 8192)
  for (int s = 0; s < steps; s += 2) {
    int k1 = (s + 1) << 5, k2 = (s + 2) << 5;
    __builtin_prefetch((const void*)(Arow + ((s + 4) << 5)), 0, 1);
    v16bf a1 = load_frag_a(Arow + k1, lda);
    v16bf b1[JF];
#pragma unroll
    for (int j = 0; j < JF; ++j) b1[j] = load_frag_bT(Bbase + (i64)j * 16 * ldbt + k1, ldbt);
#pragma unroll
    for (int j = 0; j < JF; ++j)
      acc[j] = __builtin_amdgcn_wmma_f32_16x16x32_bf16(
          false, a0, false, b0[j], (short)0, acc[j], false, false);
    a0 = load_frag_a(Arow + k2, lda);  // final iter overreads into pad (safe)
#pragma unroll
    for (int j = 0; j < JF; ++j) b0[j] = load_frag_bT(Bbase + (i64)j * 16 * ldbt + k2, ldbt);
#pragma unroll
    for (int j = 0; j < JF; ++j)
      acc[j] = __builtin_amdgcn_wmma_f32_16x16x32_bf16(
          false, a1, false, b1[j], (short)0, acc[j], false, false);
  }
#pragma unroll
  for (int j = 0; j < JF; ++j)
    store_frag_c(C + (i64)row0 * ldc + col0 + j * 16, ldc, acc[j], bias, col0 + j * 16);
}

// ---------------- adj: rowsum -> 1/(sum+1), convert to bf16 ----------------
__global__ void adj_prep(const float* __restrict__ adj, __bf16* __restrict__ adjb,
                         float* __restrict__ inv_rs) {
  int m = blockIdx.x;
  const float* row = adj + (i64)m * NN;
  __bf16* orow = adjb + (i64)m * NN;
  float s = 0.0f;
  for (int n = threadIdx.x; n < NN; n += 256) {
    float v = row[n];
    s += v;
    orow[n] = (__bf16)v;
  }
  __shared__ float sd[256];
  sd[threadIdx.x] = s; __syncthreads();
#pragma unroll
  for (int o = 128; o > 0; o >>= 1) {
    if (threadIdx.x < o) sd[threadIdx.x] += sd[threadIdx.x + o];
    __syncthreads();
  }
  if (threadIdx.x == 0) inv_rs[m] = 1.0f / (sd[0] + 1.0f);  // +1 from identity
}

// ---------------- x [B,128,N] f32 -> xt [B*N,128] bf16 (LDS tile transpose) ----------------
__global__ void transpose_x(const float* __restrict__ x, __bf16* __restrict__ xt) {
  __shared__ float tile[32][33];
  int b = blockIdx.z, n0 = blockIdx.x * 32, c0 = blockIdx.y * 32;
  int tx = threadIdx.x, ty = threadIdx.y;
#pragma unroll
  for (int i = 0; i < 32; i += 8)
    tile[ty + i][tx] = x[((i64)b * HID + c0 + ty + i) * NN + n0 + tx];
  __syncthreads();
#pragma unroll
  for (int i = 0; i < 32; i += 8)
    xt[((i64)b * NN + n0 + ty + i) * HID + c0 + tx] = (__bf16)tile[tx][ty + i];
}

// ---------------- tmp [B*N,128] f32 -> out [B,128,N] f32 ----------------
__global__ void transpose_out(const float* __restrict__ t, float* __restrict__ out) {
  __shared__ float tile[32][33];
  int b = blockIdx.z, n0 = blockIdx.x * 32, c0 = blockIdx.y * 32;
  int tx = threadIdx.x, ty = threadIdx.y;
#pragma unroll
  for (int i = 0; i < 32; i += 8)
    tile[ty + i][tx] = t[((i64)b * NN + n0 + ty + i) * HID + c0 + tx];  // [n][c]
  __syncthreads();
#pragma unroll
  for (int i = 0; i < 32; i += 8)
    out[((i64)b * HID + c0 + ty + i) * NN + n0 + tx] = tile[tx][ty + i];
}

__global__ void f32_to_bf16(const float* __restrict__ in, __bf16* __restrict__ out, i64 n) {
  i64 i = (i64)blockIdx.x * 256 + threadIdx.x;
  if (i < n) out[i] = (__bf16)in[i];
}

// kvs [bh, m, d] f32 -> kvsT [bh, d, m] bf16   (tiny: 16*128*128)
__global__ void kvs_convT(const float* __restrict__ in, __bf16* __restrict__ out) {
  int t = blockIdx.x * 256 + threadIdx.x;
  int bh = t >> 14, r = t & 16383, m = r >> 7, d = r & 127;
  out[(i64)bh * 16384 + d * 128 + m] = (__bf16)in[t];
}

__device__ inline float block_reduce128(float v, float* sd) {
  int d = threadIdx.x;
  sd[d] = v; __syncthreads();
#pragma unroll
  for (int o = 64; o > 0; o >>= 1) {
    if (d < o) sd[d] += sd[d + o];
    __syncthreads();
  }
  float r = sd[0];
  __syncthreads();
  return r;
}

// ---------------- LayerNorm (+optional relu), write f32 + bf16 ----------------
__global__ void ln_act(const float* __restrict__ in, const float* __restrict__ g,
                       const float* __restrict__ bta, int relu,
                       float* __restrict__ prev, __bf16* __restrict__ prevb) {
  i64 row = blockIdx.x;
  int d = threadIdx.x;
  __shared__ float sd[128];
  float v = in[row * HID + d];
  float m = block_reduce128(v, sd) * (1.0f / HID);
  float c = v - m;
  float var = block_reduce128(c * c, sd) * (1.0f / HID);
  float y = c * rsqrtf(var + 1e-5f) * g[d] + bta[d];
  if (relu) y = fmaxf(y, 0.0f);
  prev[row * HID + d] = y;
  prevb[row * HID + d] = (__bf16)y;
}

// ---------------- per-(b,n): L2-normalize q,k per head; reshuffle v ----------------
// Writes: qhat row-major [bh,n,d]; khatT and vT transposed [bh,d,n] (GEMM operands);
//         vbar f32 [b,n,d]; vbarT bf16 [(b*128+d), n] (GCN B-operand, transposed).
__global__ void qk_norm(const float* __restrict__ q, const float* __restrict__ k,
                        const float* __restrict__ v, __bf16* __restrict__ qhat,
                        __bf16* __restrict__ khatT, __bf16* __restrict__ vT,
                        float* __restrict__ vbar, __bf16* __restrict__ vbarT) {
  i64 bn = blockIdx.x;
  int b = (int)(bn >> 13), n = (int)(bn & (NN - 1));
  int d = threadIdx.x;
  __shared__ float sd[128];
  float vacc = 0.0f;
#pragma unroll
  for (int h = 0; h < NH; ++h) {
    i64 base = bn * HD + h * HID + d;
    float qv = q[base], kv = k[base], vv = v[base];
    float qn = rsqrtf(block_reduce128(qv * qv, sd));
    float kn = rsqrtf(block_reduce128(kv * kv, sd));
    i64 bh = (i64)b * NH + h;
    qhat[(bh * NN + n) * HID + d] = (__bf16)(qv * qn);
    khatT[(bh * HID + d) * NN + n] = (__bf16)(kv * kn);
    vT[(bh * HID + d) * NN + n] = (__bf16)vv;
    vacc += vv;
  }
  float vm = vacc * (1.0f / NH);
  vbar[bn * HID + d] = vm;
  vbarT[((i64)b * HID + d) * NN + n] = (__bf16)vm;
}

// ---------------- ksum[b,h,d] = sum_n khat ; vsum[b,h,d] = sum_n v ----------------
__global__ void reduce_sums(const __bf16* __restrict__ khatT, const __bf16* __restrict__ vT,
                            float* __restrict__ ksum, float* __restrict__ vsum) {
  int bh = blockIdx.x, d = threadIdx.x;
  float ks = 0.0f, vs = 0.0f;
  const __bf16* krow = khatT + ((i64)bh * HID + d) * NN;
  const __bf16* vrow = vT + ((i64)bh * HID + d) * NN;
  for (int n = 0; n < NN; ++n) {
    ks += (float)krow[n];
    vs += (float)vrow[n];
  }
  ksum[bh * HID + d] = ks;
  vsum[bh * HID + d] = vs;
}

// ---------------- fused attention combine + GCN epilogue + residual + LN ----------------
__global__ void combine(const float* __restrict__ num, const float* __restrict__ vsum,
                        const __bf16* __restrict__ qhat, const float* __restrict__ ksum,
                        const float* __restrict__ G, const float* __restrict__ vbar,
                        const float* __restrict__ inv_rs, const float* __restrict__ prevIn,
                        const float* __restrict__ g, const float* __restrict__ bta,
                        float* __restrict__ prevOut, __bf16* __restrict__ prevbOut) {
  i64 bn = blockIdx.x;
  int b = (int)(bn >> 13), n = (int)(bn & (NN - 1));
  int d = threadIdx.x;
  __shared__ float sd[128];
  __shared__ float qks[NH];
#pragma unroll
  for (int h = 0; h < NH; ++h) {
    i64 bh = (i64)b * NH + h;
    float qv = (float)qhat[(bh * NN + n) * HID + d];
    float s = block_reduce128(qv * ksum[bh * HID + d], sd);
    if (d == 0) qks[h] = s;
  }
  __syncthreads();
  float acc = 0.0f;
#pragma unroll
  for (int h = 0; h < NH; ++h) {
    i64 bh = (i64)b * NH + h;
    float nu = num[(bh * NN + n) * HID + d] + vsum[bh * HID + d];
    acc += nu / (qks[h] + (float)NN);
  }
  acc *= (1.0f / NH);
  float gc = (G[(i64)n * HD + b * HID + d] + vbar[bn * HID + d]) * inv_rs[n];
  float y = 0.5f * (acc + gc) + 0.5f * prevIn[bn * HID + d];
  float m = block_reduce128(y, sd) * (1.0f / HID);
  float c = y - m;
  float var = block_reduce128(c * c, sd) * (1.0f / HID);
  float o = c * rsqrtf(var + 1e-5f) * g[d] + bta[d];
  prevOut[bn * HID + d] = o;
  prevbOut[bn * HID + d] = (__bf16)o;
}

// =====================================================================================
extern "C" void kernel_launch(void* const* d_in, const int* in_sizes, int n_in,
                              void* d_out, int out_size, void* d_ws, size_t ws_size,
                              hipStream_t stream) {
  const float* x     = (const float*)d_in[0];
  const float* adj   = (const float*)d_in[1];
  const float* fc0_w = (const float*)d_in[2];
  const float* fc0_b = (const float*)d_in[3];
  const float* fc1_w = (const float*)d_in[4];
  const float* fc1_b = (const float*)d_in[5];
  const float* ln0_g = (const float*)d_in[6];
  const float* ln0_b = (const float*)d_in[7];
  float* out = (float*)d_out;

  char* ws = (char*)d_ws;
  size_t off = 0;
  auto alloc = [&](size_t bytes) { void* p = ws + off; off = (off + bytes + 255) & ~(size_t)255; return p; };
  __bf16* adjb   = (__bf16*)alloc((size_t)NN * NN * 2);        // 128 MB (L2-resident)
  float*  inv_rs = (float*) alloc((size_t)NN * 4);
  float*  prevf  = (float*) alloc((size_t)BZ * NN * HID * 4);
  __bf16* prevb  = (__bf16*)alloc((size_t)BZ * NN * HID * 2);  // also xt buffer
  __bf16* w0B    = (__bf16*)alloc((size_t)HID * HID * 2);      // fc0_w bf16 (already BT layout)
  __bf16* w1B    = (__bf16*)alloc((size_t)HID * HID * 2);
  __bf16* wqB    = (__bf16*)alloc((size_t)HD * HID * 2);
  __bf16* wkB    = (__bf16*)alloc((size_t)HD * HID * 2);
  __bf16* wvB    = (__bf16*)alloc((size_t)HD * HID * 2);
  float*  qf     = (float*) alloc((size_t)BZ * NN * HD * 4);   // aliases: fc0-out, num
  float*  kf     = (float*) alloc((size_t)BZ * NN * HD * 4);   // aliases: G, fc1 tmp
  float*  vf     = (float*) alloc((size_t)BZ * NN * HD * 4);
  __bf16* qhat   = (__bf16*)alloc((size_t)BZ * NH * NN * HID * 2);
  __bf16* khatT  = (__bf16*)alloc((size_t)BZ * NH * HID * NN * 2);
  __bf16* vTb    = (__bf16*)alloc((size_t)BZ * NH * HID * NN * 2);
  float*  vbarf  = (float*) alloc((size_t)BZ * NN * HID * 4);
  __bf16* vbarT  = (__bf16*)alloc((size_t)BZ * HID * NN * 2);  // [(b*128+d), n]
  float*  ksum   = (float*) alloc((size_t)BZ * NH * HID * 4);
  float*  vsum   = (float*) alloc((size_t)BZ * NH * HID * 4);
  float*  kvsf   = (float*) alloc((size_t)BZ * NH * HID * HID * 4);
  __bf16* kvsbT  = (__bf16*)alloc((size_t)BZ * NH * HID * HID * 2);
  (void)alloc(4096);  // pad: GEMM pipeline preload may overread one 64B k-step
  float*  numf   = qf;
  float*  Gf     = kf;
  float*  tmpf   = kf;
  (void)ws_size; (void)n_in; (void)in_sizes; (void)out_size;

  const int M = BZ * NN;  // 32768 node-feature rows

  adj_prep<<<NN, 256, 0, stream>>>(adj, adjb, inv_rs);
  transpose_x<<<dim3(NN / 32, HID / 32, BZ), dim3(32, 8), 0, stream>>>(x, prevb);
  f32_to_bf16<<<(HID * HID + 255) / 256, 256, 0, stream>>>(fc0_w, w0B, HID * HID);
  f32_to_bf16<<<(HID * HID + 255) / 256, 256, 0, stream>>>(fc1_w, w1B, HID * HID);

  // fc0: [M,128] @ W^T, W given as BT[128,128]
  gemm_bf16<4><<<dim3(M / 64, 2, 1), 128, 0, stream>>>(
      prevb, w0B, qf, fc0_b, M, HID, HID, HID, HID, HID, 0, 0, 0);
  ln_act<<<M, 128, 0, stream>>>(qf, ln0_g, ln0_b, 1, prevf, prevb);

  for (int i = 0; i < 2; ++i) {
    const float* wq_w = (const float*)d_in[12 + i * 6 + 0];
    const float* wq_b = (const float*)d_in[12 + i * 6 + 1];
    const float* wk_w = (const float*)d_in[12 + i * 6 + 2];
    const float* wk_b = (const float*)d_in[12 + i * 6 + 3];
    const float* wv_w = (const float*)d_in[12 + i * 6 + 4];
    const float* wv_b = (const float*)d_in[12 + i * 6 + 5];
    const float* lng  = (const float*)d_in[8 + 2 * i];
    const float* lnb  = (const float*)d_in[9 + 2 * i];

    f32_to_bf16<<<(HD * HID + 255) / 256, 256, 0, stream>>>(wq_w, wqB, HD * HID);
    f32_to_bf16<<<(HD * HID + 255) / 256, 256, 0, stream>>>(wk_w, wkB, HD * HID);
    f32_to_bf16<<<(HD * HID + 255) / 256, 256, 0, stream>>>(wv_w, wvB, HD * HID);

    // Q/K/V projections: [M,128] @ [128,512], BT = weight [512,128]
    gemm_bf16<8><<<dim3(M / 64, HD / 128, 1), 128, 0, stream>>>(
        prevb, wqB, qf, wq_b, M, HD, HID, HID, HID, HD, 0, 0, 0);
    gemm_bf16<8><<<dim3(M / 64, HD / 128, 1), 128, 0, stream>>>(
        prevb, wkB, kf, wk_b, M, HD, HID, HID, HID, HD, 0, 0, 0);
    gemm_bf16<8><<<dim3(M / 64, HD / 128, 1), 128, 0, stream>>>(
        prevb, wvB, vf, wv_b, M, HD, HID, HID, HID, HD, 0, 0, 0);

    qk_norm<<<M, 128, 0, stream>>>(qf, kf, vf, qhat, khatT, vTb, vbarf, vbarT);
    reduce_sums<<<BZ * NH, 128, 0, stream>>>(khatT, vTb, ksum, vsum);

    // kvs[b,h] = khat^T @ v : batch 16, [128,8192]@[8192,128]; BT = vT [d,n]
    gemm_bf16<4><<<dim3(2, 2, BZ * NH), 128, 0, stream>>>(
        khatT, vTb, kvsf, nullptr, HID, HID, NN, NN, NN, HID,
        (i64)HID * NN, (i64)HID * NN, (i64)HID * HID);
    kvs_convT<<<(BZ * NH * HID * HID) / 256, 256, 0, stream>>>(kvsf, kvsbT);

    // num[b,h] = qhat @ kvs : batch 16, [8192,128]@[128,128]; BT = kvs^T [d,m]
    gemm_bf16<4><<<dim3(NN / 64, 2, BZ * NH), 128, 0, stream>>>(
        qhat, kvsbT, numf, nullptr, NN, HID, HID, HID, HID, HID,
        (i64)NN * HID, (i64)HID * HID, (i64)NN * HID);

    // GCN: G = adj @ vbar_cat : [8192,8192]@[8192,512]; BT = vbarT [(b*128+d), n]
    gemm_bf16<8><<<dim3(NN / 64, HD / 128, 1), 128, 0, stream>>>(
        adjb, vbarT, Gf, nullptr, NN, HD, NN, NN, NN, HD, 0, 0, 0);

    combine<<<M, 128, 0, stream>>>(numf, vsum, qhat, ksum, Gf, vbarf, inv_rs,
                                   prevf, lng, lnb, prevf, prevb);
  }

  gemm_bf16<4><<<dim3(M / 64, 2, 1), 128, 0, stream>>>(
      prevb, w1B, tmpf, fc1_b, M, HID, HID, HID, HID, HID, 0, 0, 0);
  transpose_out<<<dim3(NN / 32, HID / 32, BZ), dim3(32, 8), 0, stream>>>(tmpf, out);
}
